// SConv_xyz_79319456023047
// MI455X (gfx1250) — matrix operand order repaired and checked by reading, not compiled, not run on hardware
//
#include <hip/hip_runtime.h>
#include <hip/hip_bf16.h>
#include <math.h>

// ---------------------------------------------------------------------------
// MI455X (gfx1250) implementation of the DCN feature-alignment block.
// All GEMM-like work (conv3x3, 1x1 conv, both deformable convs) runs on
// v_wmma_f32_16x16x32_bf16: weights pre-packed to a GEMM bf16 layout in ws,
// im2col / bilinear-sampled activations staged per-tile in LDS as bf16.
// LDS row stride is padded (+8 bf16) so fragment reads/writes are
// bank-conflict-free (row n -> banks 4n..4n+3).
// ---------------------------------------------------------------------------

typedef __attribute__((ext_vector_type(16))) __bf16 bf16x16;
typedef __attribute__((ext_vector_type(8)))  __bf16 bf16x8;
typedef __attribute__((ext_vector_type(8)))  float  f32x8;

#define MODE_DIRECT3 0
#define MODE_BILIN   1
#define MODE_PT1     2

#define EPI_NONE       0
#define EPI_BIAS       1
#define EPI_SCALE_RELU 2
#define EPI_SIGMOID    3

// ---------------------------------------------------------------------------
// Bilinear resize, align_corners=True  (S: [B,C,128,128] -> [B,C,64,64])
// ---------------------------------------------------------------------------
__global__ __launch_bounds__(256) void resize_ac_kernel(
    const float* __restrict__ S, float* __restrict__ Si,
    int BC, int Hin, int Win, int Ho, int Wo)
{
    int i = blockIdx.x * 256 + threadIdx.x;
    int total = BC * Ho * Wo;
    if (i >= total) return;
    int x  = i % Wo;
    int y  = (i / Wo) % Ho;
    int bc = i / (Wo * Ho);
    float sy = (float)(Hin - 1) / (float)(Ho - 1);
    float sx = (float)(Win - 1) / (float)(Wo - 1);
    float py = y * sy, px = x * sx;
    float y0f = floorf(py), x0f = floorf(px);
    int y0 = (int)y0f, x0 = (int)x0f;
    int y1 = min(y0 + 1, Hin - 1);
    int x1 = min(x0 + 1, Win - 1);
    float wy = py - y0f, wx = px - x0f;
    const float* base = S + (size_t)bc * Hin * Win;
    float v00 = base[y0 * Win + x0], v01 = base[y0 * Win + x1];
    float v10 = base[y1 * Win + x0], v11 = base[y1 * Win + x1];
    float top = v00 * (1.f - wx) + v01 * wx;
    float bot = v10 * (1.f - wx) + v11 * wx;
    Si[i] = top * (1.f - wy) + bot * wy;
}

// ---------------------------------------------------------------------------
// Pack conv weights [O][C][TAPS] fp32 -> GEMM bf16 [OPAD][KPAD], k = tap*C + c
// (zero-padded in both O and K).
// ---------------------------------------------------------------------------
__global__ __launch_bounds__(256) void prep_w_kernel(
    const float* __restrict__ w, __bf16* __restrict__ wp,
    int O, int C, int TAPS, int OPAD, int KPAD)
{
    int i = blockIdx.x * 256 + threadIdx.x;
    int total = OPAD * KPAD;
    if (i >= total) return;
    int o = i / KPAD, k = i % KPAD;
    float v = 0.f;
    if (o < O && k < C * TAPS) {
        int tap = k / C, c = k % C;
        v = w[((size_t)o * C + c) * TAPS + tap];
    }
    wp[i] = (__bf16)v;
}

// ---------------------------------------------------------------------------
// Unified implicit-GEMM conv / DCN kernel.
//   D[m,n] = sum_k A[m,k] * B[k,n]
//   A = packed weights (bf16, global, L2/WGP$ resident)
//   B = im2col / bilinearly-sampled activations (bf16, staged in LDS)
// 256 threads = 8 waves; wave (mw,nw) computes MT_PER_WAVE 16x16 tiles.
// ---------------------------------------------------------------------------
template<int CIN, int COUT, int COUT_PAD, int TAPS, int MODE, int EPI,
         int N_WAVES, int MT_PER_WAVE, bool HAS_MASK>
__global__ __launch_bounds__(256, 1)
void gemm_conv_kernel(const float* __restrict__ in,
                      const __bf16* __restrict__ wp,
                      const float* __restrict__ offs,
                      const float* __restrict__ mask,
                      const float* __restrict__ bias,
                      const float* __restrict__ scale,
                      const float* __restrict__ shift,
                      float* __restrict__ out,
                      int H, int W)
{
    constexpr int N_TILE  = N_WAVES * 16;
    constexpr int M_WAVES = 8 / N_WAVES;
    constexpr int K_TOT   = TAPS * CIN;
    constexpr int K_PAD   = (K_TOT + 31) & ~31;
    constexpr int K_ROW   = K_PAD + 8;   // +16B: row n -> banks 4n..4n+3
    static_assert(M_WAVES * MT_PER_WAVE * 16 == COUT_PAD, "bad tiling");

    __shared__ __align__(32) __bf16 sB[N_TILE * K_ROW];
    constexpr int NPAR = (MODE == MODE_BILIN) ? N_TILE * TAPS : 1;
    __shared__ int   sY0[NPAR];
    __shared__ int   sX0[NPAR];
    __shared__ float sCW[4][NPAR];

    const int tid = threadIdx.x;
    const int HW = H * W;
    const int tile_p0 = blockIdx.x * N_TILE;

    // ---- phase 0: per-(pixel,tap) bilinear params (DCN only) ----
    if constexpr (MODE == MODE_BILIN) {
        for (int idx = tid; idx < N_TILE * TAPS; idx += 256) {
            const int n = idx / TAPS, tap = idx % TAPS;
            const int p = tile_p0 + n;
            const int b = p / HW, r = p % HW, yy = r / W, xx = r % W;
            const float dyo = offs[((size_t)b * (2 * TAPS) + 2 * tap + 0) * HW + r];
            const float dxo = offs[((size_t)b * (2 * TAPS) + 2 * tap + 1) * HW + r];
            const float py = (float)(yy + tap / 3 - 1) + dyo;
            const float px = (float)(xx + tap % 3 - 1) + dxo;
            const float y0f = floorf(py), x0f = floorf(px);
            const int y0 = (int)y0f, x0 = (int)x0f;
            const float fy = py - y0f, fx = px - x0f;
            float mm = 1.0f;
            if constexpr (HAS_MASK)
                mm = mask[((size_t)b * TAPS + tap) * HW + r];
            const float vy0 = (y0 >= 0 && y0 <= H - 1) ? 1.f : 0.f;
            const float vy1 = (y0 + 1 >= 0 && y0 + 1 <= H - 1) ? 1.f : 0.f;
            const float vx0 = (x0 >= 0 && x0 <= W - 1) ? 1.f : 0.f;
            const float vx1 = (x0 + 1 >= 0 && x0 + 1 <= W - 1) ? 1.f : 0.f;
            sY0[idx] = y0;
            sX0[idx] = x0;
            sCW[0][idx] = (1.f - fy) * (1.f - fx) * vy0 * vx0 * mm;
            sCW[1][idx] = (1.f - fy) * fx         * vy0 * vx1 * mm;
            sCW[2][idx] = fy * (1.f - fx)         * vy1 * vx0 * mm;
            sCW[3][idx] = fy * fx                 * vy1 * vx1 * mm;
        }
        __syncthreads();
    }

    // ---- phase 1: stage B[k,n] into LDS as bf16 (layout: sB[n*K_ROW + k]).
    // Pixel-fastest mapping: consecutive lanes -> consecutive pixels of one
    // channel (near-coalesced global gathers); padded K_ROW keeps the LDS
    // writes (stride 4n banks) conflict-free.
    for (int idx = tid; idx < N_TILE * CIN; idx += 256) {
        const int n = idx % N_TILE, c = idx / N_TILE;
        const int p = tile_p0 + n;
        const int b = p / HW, r = p % HW, yy = r / W, xx = r % W;
        const float* __restrict__ base = in + ((size_t)b * CIN + c) * HW;
        if constexpr (MODE == MODE_PT1) {
            sB[n * K_ROW + c] = (__bf16)base[r];
        } else if constexpr (MODE == MODE_DIRECT3) {
            #pragma unroll
            for (int tap = 0; tap < 9; ++tap) {
                const int iy = yy + tap / 3 - 1, ix = xx + tap % 3 - 1;
                const float v = (iy >= 0 && iy < H && ix >= 0 && ix < W)
                                    ? base[iy * W + ix] : 0.f;
                sB[n * K_ROW + tap * CIN + c] = (__bf16)v;
            }
        } else { // MODE_BILIN
            #pragma unroll
            for (int tap = 0; tap < 9; ++tap) {
                const int j = n * TAPS + tap;
                const int y0 = sY0[j], x0 = sX0[j];
                const int yc0 = min(max(y0, 0), H - 1);
                const int yc1 = min(max(y0 + 1, 0), H - 1);
                const int xc0 = min(max(x0, 0), W - 1);
                const int xc1 = min(max(x0 + 1, 0), W - 1);
                const float g = sCW[0][j] * base[yc0 * W + xc0]
                              + sCW[1][j] * base[yc0 * W + xc1]
                              + sCW[2][j] * base[yc1 * W + xc0]
                              + sCW[3][j] * base[yc1 * W + xc1];
                sB[n * K_ROW + tap * CIN + c] = (__bf16)g;
            }
        }
    }
    if constexpr (K_PAD > K_TOT) {
        constexpr int PADK = K_PAD - K_TOT;
        for (int idx = tid; idx < N_TILE * PADK; idx += 256) {
            const int n = idx % N_TILE, j = idx / N_TILE;
            sB[n * K_ROW + K_TOT + j] = (__bf16)0.f;
        }
    }
    __syncthreads();

    // ---- phase 2: WMMA main loop ----
    const int wave = tid >> 5, lane = tid & 31;
    const int mw = wave % M_WAVES, nw = wave / M_WAVES;
    const int half = lane >> 4, ncol = lane & 15;

    f32x8 acc[MT_PER_WAVE] = {};

    // B fragment (32x16 KxN, bf16): n = lane&15; lanes 0-15 hold K=0..15,
    // lanes 16-31 hold K=16..31; k contiguous per lane -> 2x16B LDS reads.
    const __bf16* __restrict__ srow = &sB[(nw * 16 + ncol) * K_ROW];

    for (int ks = 0; ks < K_PAD; ks += 32) {
        const bf16x8 blo = *(const bf16x8*)(srow + ks + half * 16);
        const bf16x8 bhi = *(const bf16x8*)(srow + ks + half * 16 + 8);
        const bf16x16 bfrag = __builtin_shufflevector(
            blo, bhi, 0, 1, 2, 3, 4, 5, 6, 7, 8, 9, 10, 11, 12, 13, 14, 15);
        #pragma unroll
        for (int t = 0; t < MT_PER_WAVE; ++t) {
            // A fragment (16x32 MxK, bf16): m = lane&15; lanes 0-15 hold
            // k = 0..15, lanes 16-31 hold k = 16..31 (matches B).
            const int m = (mw * MT_PER_WAVE + t) * 16 + ncol;
            const __bf16* __restrict__ wrow = wp + (size_t)m * K_PAD + ks;
            const bf16x8 alo = *(const bf16x8*)(wrow + half * 16);
            const bf16x8 ahi = *(const bf16x8*)(wrow + half * 16 + 8);
            const bf16x16 afrag = __builtin_shufflevector(
                alo, ahi, 0, 1, 2, 3, 4, 5, 6, 7, 8, 9, 10, 11, 12, 13, 14, 15);
            acc[t] = __builtin_amdgcn_wmma_f32_16x16x32_bf16(
                false, afrag, false, bfrag, (short)0, acc[t], false, false);
        }
    }

    // ---- epilogue: C/D layout: n = lane&15, m = half*8 + vgpr ----
    const int p = tile_p0 + nw * 16 + ncol;
    const int b = p / HW, r = p % HW;
    #pragma unroll
    for (int t = 0; t < MT_PER_WAVE; ++t) {
        const int mbase = (mw * MT_PER_WAVE + t) * 16 + half * 8;
        #pragma unroll
        for (int rr = 0; rr < 8; ++rr) {
            const int m = mbase + rr;
            const int mc = m < COUT ? m : 0;
            float v = acc[t][rr];
            if constexpr (EPI == EPI_BIAS) {
                v += bias[mc];
            } else if constexpr (EPI == EPI_SCALE_RELU) {
                v = (v + bias[mc]) * scale[mc] + shift[mc];
                v = v > 0.f ? v : 0.f;
            } else if constexpr (EPI == EPI_SIGMOID) {
                v = 1.f / (1.f + __expf(-(v + bias[mc])));
            }
            if (m < COUT)
                out[((size_t)b * COUT + m) * HW + r] = v;
        }
    }
}

// ---------------------------------------------------------------------------
// Host-side orchestration
// ---------------------------------------------------------------------------
extern "C" void kernel_launch(void* const* d_in, const int* in_sizes, int n_in,
                              void* d_out, int out_size, void* d_ws, size_t ws_size,
                              hipStream_t stream)
{
    const float* x      = (const float*)d_in[0];   // [8,256,64,64]
    const float* S      = (const float*)d_in[1];   // [8,3,128,128]
    const float* sp_w1  = (const float*)d_in[2];
    const float* sp_b1  = (const float*)d_in[3];
    const float* sp_s1  = (const float*)d_in[4];
    const float* sp_o1  = (const float*)d_in[5];
    const float* sp_w2  = (const float*)d_in[6];
    const float* sp_b2  = (const float*)d_in[7];
    const float* sp_s2  = (const float*)d_in[8];
    const float* sp_o2  = (const float*)d_in[9];
    const float* sp_w3  = (const float*)d_in[10];
    const float* sp_b3  = (const float*)d_in[11];
    const float* sp_s3  = (const float*)d_in[12];
    const float* sp_o3  = (const float*)d_in[13];
    const float* off_w  = (const float*)d_in[14];
    const float* off_b  = (const float*)d_in[15];
    const float* dcow   = (const float*)d_in[16];
    const float* dcob   = (const float*)d_in[17];
    const float* dc_w   = (const float*)d_in[18];
    const float* m_w    = (const float*)d_in[19];
    const float* m_b    = (const float*)d_in[20];
    const float* weight = (const float*)d_in[21];
    const float* bias   = (const float*)d_in[22];
    float* outp = (float*)d_out;

    const int B = 8, H = 64, W = 64, HW = H * W;
    const int NPIX = B * HW; // 32768

    // workspace carve-out (deterministic, pure pointer math)
    char* ws = (char*)d_ws;
    size_t off = 0;
    auto carve = [&](size_t bytes) -> char* {
        char* p = ws + off;
        off = (off + bytes + 255) & ~(size_t)255;
        return p;
    };
    float*  Si      = (float*)carve((size_t)B * 3 * HW * 4);
    float*  h1      = (float*)carve((size_t)B * 64 * HW * 4);
    float*  h2      = (float*)carve((size_t)B * 64 * HW * 4);
    float*  h3      = (float*)carve((size_t)B * 64 * HW * 4);
    float*  offbuf  = (float*)carve((size_t)B * 18 * HW * 4);
    float*  dcoffb  = (float*)carve((size_t)B * 18 * HW * 4);
    float*  featb   = (float*)carve((size_t)B * 64 * HW * 4);
    float*  maskb   = (float*)carve((size_t)B * 9 * HW * 4);
    __bf16* wp1     = (__bf16*)carve((size_t)64 * 32 * 2);     // 3->64, K=27 pad 32
    __bf16* wp2     = (__bf16*)carve((size_t)64 * 576 * 2);
    __bf16* wp3     = (__bf16*)carve((size_t)64 * 576 * 2);
    __bf16* wpoff   = (__bf16*)carve((size_t)32 * 576 * 2);    // 18 pad 32
    __bf16* wpdcoff = (__bf16*)carve((size_t)32 * 576 * 2);
    __bf16* wpdc    = (__bf16*)carve((size_t)64 * 576 * 2);
    __bf16* wpm     = (__bf16*)carve((size_t)16 * 64 * 2);     // 9 pad 16
    __bf16* wpfin   = (__bf16*)carve((size_t)256 * 2304 * 2);
    (void)ws_size; (void)in_sizes; (void)n_in; (void)out_size;

    // 1) bilinear resize S -> Si
    {
        int total = B * 3 * HW;
        resize_ac_kernel<<<(total + 255) / 256, 256, 0, stream>>>(
            S, Si, B * 3, 128, 128, H, W);
    }

    // 2) pack all weights into bf16 GEMM layout
    auto prep = [&](const float* w, __bf16* wp, int O, int C, int T, int OP, int KP) {
        int total = OP * KP;
        prep_w_kernel<<<(total + 255) / 256, 256, 0, stream>>>(w, wp, O, C, T, OP, KP);
    };
    prep(sp_w1, wp1, 64, 3, 9, 64, 32);
    prep(sp_w2, wp2, 64, 64, 9, 64, 576);
    prep(sp_w3, wp3, 64, 64, 9, 64, 576);
    prep(off_w, wpoff, 18, 64, 9, 32, 576);
    prep(dcow, wpdcoff, 18, 64, 9, 32, 576);
    prep(dc_w, wpdc, 64, 64, 9, 64, 576);
    prep(m_w, wpm, 9, 64, 1, 16, 64);
    prep(weight, wpfin, 256, 256, 9, 256, 2304);

    // 3) spatial projector: 3x (conv3x3 + BN-fold + ReLU)
    gemm_conv_kernel<3, 64, 64, 9, MODE_DIRECT3, EPI_SCALE_RELU, 2, 1, false>
        <<<NPIX / 32, 256, 0, stream>>>(Si, wp1, nullptr, nullptr,
                                        sp_b1, sp_s1, sp_o1, h1, H, W);
    gemm_conv_kernel<64, 64, 64, 9, MODE_DIRECT3, EPI_SCALE_RELU, 2, 1, false>
        <<<NPIX / 32, 256, 0, stream>>>(h1, wp2, nullptr, nullptr,
                                        sp_b2, sp_s2, sp_o2, h2, H, W);
    gemm_conv_kernel<64, 64, 64, 9, MODE_DIRECT3, EPI_SCALE_RELU, 2, 1, false>
        <<<NPIX / 32, 256, 0, stream>>>(h2, wp3, nullptr, nullptr,
                                        sp_b3, sp_s3, sp_o3, h3, H, W);

    // 4) offset heads (conv3x3 + bias), Cout=18 padded to 32
    gemm_conv_kernel<64, 18, 32, 9, MODE_DIRECT3, EPI_BIAS, 4, 1, false>
        <<<NPIX / 64, 256, 0, stream>>>(h3, wpoff, nullptr, nullptr,
                                        off_b, nullptr, nullptr, offbuf, H, W);
    gemm_conv_kernel<64, 18, 32, 9, MODE_DIRECT3, EPI_BIAS, 4, 1, false>
        <<<NPIX / 64, 256, 0, stream>>>(h3, wpdcoff, nullptr, nullptr,
                                        dcob, nullptr, nullptr, dcoffb, H, W);

    // 5) DCNv1 on h3 -> feat (no mask, no bias)
    gemm_conv_kernel<64, 64, 64, 9, MODE_BILIN, EPI_NONE, 2, 1, false>
        <<<NPIX / 32, 256, 0, stream>>>(h3, wpdc, dcoffb, nullptr,
                                        nullptr, nullptr, nullptr, featb, H, W);

    // 6) mask head: 1x1 conv + sigmoid, Cout=9 padded to 16
    gemm_conv_kernel<64, 9, 16, 1, MODE_PT1, EPI_SIGMOID, 8, 1, false>
        <<<NPIX / 128, 256, 0, stream>>>(featb, wpm, nullptr, nullptr,
                                         m_b, nullptr, nullptr, maskb, H, W);

    // 7) final modulated DCNv2: x [8,256,64,64] -> out [8,256,64,64]
    //    N_TILE=32 (2 n-waves), 4 M-tiles/wave (M=256), K=2304 staged in
    //    ~145KB LDS (2 workgroups/WGP on 320KB) -> 2x weight reuse vs N=16.
    gemm_conv_kernel<256, 256, 256, 9, MODE_BILIN, EPI_BIAS, 2, 4, true>
        <<<NPIX / 32, 256, 0, stream>>>(x, wpfin, offbuf, maskb,
                                        bias, nullptr, nullptr, outp, H, W);
}
